// InfoNCE_Loss_86311662780620
// MI455X (gfx1250) — matrix operand order repaired and checked by reading, not compiled, not run on hardware
//
#include <hip/hip_runtime.h>
#include <hip/hip_bf16.h>
#include <math.h>
#include <stdint.h>

typedef __attribute__((ext_vector_type(16))) _Float16 v16h;
typedef __attribute__((ext_vector_type(8)))  _Float16 v8h;
typedef __attribute__((ext_vector_type(8)))  float    v8f;

#define N_FEAT 16384
#define D_FEAT 512
#define EPS_F  1e-8f
#define KSTAGE 64            // K depth staged per LDS buffer
#define NSTAGE (D_FEAT / KSTAGE)
#define LDPAD  72            // halves per LDS row (64 + 8 pad -> 144B stride, conflict-free frags)

// ---------------- helper kernels ----------------

__global__ void zero_f32_kernel(float* __restrict__ p, int n) {
    int i = blockIdx.x * blockDim.x + threadIdx.x;
    if (i < n) p[i] = 0.0f;
}

__global__ void cvt_f32_to_f16_kernel(const float* __restrict__ src,
                                      _Float16* __restrict__ dst, int n) {
    int i = blockIdx.x * blockDim.x + threadIdx.x;
    int stride = gridDim.x * blockDim.x;
    for (; i < n; i += stride) dst[i] = (_Float16)src[i];
}

// ---------------- fused GEMM + exp + row/col reduction ----------------
// Block tile: 128(M) x 128(N). 256 threads = 8 wave32, wave grid 2(M) x 4(N),
// each wave owns a 64x32 subtile = 4x2 WMMA accumulators.
// A/B tiles staged global->LDS with async copies (ASYNCcnt), double buffered.

__global__ __launch_bounds__(256) void infonce_tile_kernel(
    const _Float16* __restrict__ imgh,   // [N_FEAT, D_FEAT] row major
    const _Float16* __restrict__ txth,   // [N_FEAT, D_FEAT] row major
    const float*    __restrict__ logit_scale,
    float* __restrict__ rowsum,          // [N_FEAT] sum_j exp(s*S[i,j])
    float* __restrict__ colsum,          // [N_FEAT] sum_i exp(s*S[i,j])
    float* __restrict__ diag)            // [N_FEAT] s*S[i,i]
{
    __shared__ _Float16 As[2][128 * LDPAD];   // 2 x 18 KB
    __shared__ _Float16 Bs[2][128 * LDPAD];   // 2 x 18 KB

    const int t    = threadIdx.x;
    const int lane = t & 31;
    const int wave = t >> 5;
    const int h    = lane >> 4;          // 16-lane half select
    const int l16  = lane & 15;

    const int wm = wave & 1;             // 0..1  -> M offset wm*64
    const int wn = wave >> 1;            // 0..3  -> N offset wn*32

    const int bm = blockIdx.y * 128;
    const int bn = blockIdx.x * 128;
    const int Mb = bm + wm * 64;
    const int Nb = bn + wn * 32;

    // staging role: thread t copies row (t>>1), 4 contiguous 16B chunks
    const int srow = t >> 1;             // 0..127
    const int sseg = (t & 1) * 4;        // chunk base within the 64-half row

    const _Float16* gA = imgh + (size_t)bm * D_FEAT;
    const _Float16* gB = txth + (size_t)bn * D_FEAT;

    // LDS byte addresses (low 32 bits of the flat pointer = wave-relative LDS offset)
    uint32_t ldsA[2], ldsB[2];
    ldsA[0] = (uint32_t)(uintptr_t)(&As[0][0]);
    ldsA[1] = (uint32_t)(uintptr_t)(&As[1][0]);
    ldsB[0] = (uint32_t)(uintptr_t)(&Bs[0][0]);
    ldsB[1] = (uint32_t)(uintptr_t)(&Bs[1][0]);

    // issue one stage: 8 async 16B copies per thread (A:4 + B:4) => 32 KB/block
    auto issue_stage = [&](int ks, int buf) {
        const int k0 = ks * KSTAGE;
#pragma unroll
        for (int i = 0; i < 4; ++i) {
            const int seg = sseg + i;
            const _Float16* g = gA + (size_t)srow * D_FEAT + k0 + seg * 8;
            uint32_t l = ldsA[buf] + (uint32_t)(srow * (LDPAD * 2) + seg * 16);
            asm volatile("global_load_async_to_lds_b128 %0, %1, off"
                         :: "v"(l), "v"(g) : "memory");
        }
#pragma unroll
        for (int i = 0; i < 4; ++i) {
            const int seg = sseg + i;
            const _Float16* g = gB + (size_t)srow * D_FEAT + k0 + seg * 8;
            uint32_t l = ldsB[buf] + (uint32_t)(srow * (LDPAD * 2) + seg * 16);
            asm volatile("global_load_async_to_lds_b128 %0, %1, off"
                         :: "v"(l), "v"(g) : "memory");
        }
    };

    const v8f vzero = {};
    v8f acc[4][2];
#pragma unroll
    for (int mt = 0; mt < 4; ++mt)
#pragma unroll
        for (int nt = 0; nt < 2; ++nt) acc[mt][nt] = vzero;

    issue_stage(0, 0);

    for (int ks = 0; ks < NSTAGE; ++ks) {
        const int buf = ks & 1;
        if (ks + 1 < NSTAGE) {
            issue_stage(ks + 1, buf ^ 1);  // overwrite safe: end-barrier of ks-1 passed
            // in-order completion: oldest 8 (= stage ks) done when <= 8 outstanding
            asm volatile("s_wait_asynccnt 8" ::: "memory");
        } else {
            asm volatile("s_wait_asynccnt 0" ::: "memory");
        }
        __syncthreads();   // stage ks visible to all waves

        const _Float16* Ab = &As[buf][0];
        const _Float16* Bb = &Bs[buf][0];

#pragma unroll
        for (int ksub = 0; ksub < KSTAGE; ksub += 32) {
            v16h afrag[4];
            v16h bfrag[2];
            // A fragment (16x32): lane l16 = row; elems 0..7 = K[ksub+8h..],
            //                                  elems 8..15 = K[ksub+16+8h..]
#pragma unroll
            for (int mt = 0; mt < 4; ++mt) {
                const _Float16* p = Ab + (wm * 64 + mt * 16 + l16) * LDPAD + ksub + 8 * h;
                v8h lo = *(const v8h*)p;
                v8h hi = *(const v8h*)(p + 16);
#pragma unroll
                for (int i = 0; i < 8; ++i) { afrag[mt][i] = lo[i]; afrag[mt][i + 8] = hi[i]; }
            }
            // B fragment (32x16): lane l16 = col (= txt row); elems 0..7 = K[ksub+8h..],
            //                                                elems 8..15 = K[ksub+16+8h..]
#pragma unroll
            for (int nt = 0; nt < 2; ++nt) {
                const _Float16* q = Bb + (wn * 32 + nt * 16 + l16) * LDPAD + ksub + 8 * h;
                v8h lo = *(const v8h*)q;
                v8h hi = *(const v8h*)(q + 16);
#pragma unroll
                for (int i = 0; i < 8; ++i) { bfrag[nt][i] = lo[i]; bfrag[nt][i + 8] = hi[i]; }
            }

#pragma unroll
            for (int mt = 0; mt < 4; ++mt)
#pragma unroll
                for (int nt = 0; nt < 2; ++nt)
                    acc[mt][nt] = __builtin_amdgcn_wmma_f32_16x16x32_f16(
                        false, afrag[mt], false, bfrag[nt],
                        (short)0, acc[mt][nt], false, false);
        }
        __syncthreads();   // all waves done reading buf before it is overwritten
    }

    const float s = *logit_scale;
    const bool hasDiag = (blockIdx.x == blockIdx.y);

    // scale -> capture diagonal -> exponentiate (in place)
#pragma unroll
    for (int mt = 0; mt < 4; ++mt) {
#pragma unroll
        for (int nt = 0; nt < 2; ++nt) {
#pragma unroll
            for (int r = 0; r < 8; ++r) {
                float v = s * acc[mt][nt][r];
                if (hasDiag) {
                    int gm = Mb + mt * 16 + 8 * h + r;   // C/D layout: VGPR r -> row 8h+r
                    int gn = Nb + nt * 16 + l16;         // lane -> column
                    if (gm == gn) diag[gm] = v;          // single writer
                }
                acc[mt][nt][r] = __expf(v);
            }
        }
    }

    // row sums: reduce over this wave's 32 columns (lanes within each half)
#pragma unroll
    for (int mt = 0; mt < 4; ++mt) {
#pragma unroll
        for (int r = 0; r < 8; ++r) {
            float v = acc[mt][0][r] + acc[mt][1][r];
            v += __shfl_xor(v, 1, 32);
            v += __shfl_xor(v, 2, 32);
            v += __shfl_xor(v, 4, 32);
            v += __shfl_xor(v, 8, 32);
            if (l16 == 0) {
                int gm = Mb + mt * 16 + 8 * h + r;
                atomicAdd(&rowsum[gm], v);
            }
        }
    }

    // col sums: each lane owns one column per N-tile; combine the two halves
#pragma unroll
    for (int nt = 0; nt < 2; ++nt) {
        float c = 0.0f;
#pragma unroll
        for (int mt = 0; mt < 4; ++mt)
#pragma unroll
            for (int r = 0; r < 8; ++r) c += acc[mt][nt][r];
        c += __shfl_xor(c, 16, 32);
        if (h == 0) {
            int gn = Nb + nt * 16 + l16;
            atomicAdd(&colsum[gn], c);
        }
    }
}

// ---------------- finalize: loss = mean_i [0.5*(log(rs+eps)+log(cs+eps)) - diag] ----

__global__ void infonce_finalize_kernel(const float* __restrict__ rowsum,
                                        const float* __restrict__ colsum,
                                        const float* __restrict__ diag,
                                        float* __restrict__ out) {
    __shared__ float red[256];
    float acc = 0.0f;
    for (int i = threadIdx.x; i < N_FEAT; i += 256) {
        acc += 0.5f * (logf(rowsum[i] + EPS_F) + logf(colsum[i] + EPS_F)) - diag[i];
    }
    red[threadIdx.x] = acc;
    __syncthreads();
    for (int sft = 128; sft > 0; sft >>= 1) {
        if ((int)threadIdx.x < sft) red[threadIdx.x] += red[threadIdx.x + sft];
        __syncthreads();
    }
    if (threadIdx.x == 0) out[0] = red[0] / (float)N_FEAT;
}

// ---------------- launch ----------------

extern "C" void kernel_launch(void* const* d_in, const int* in_sizes, int n_in,
                              void* d_out, int out_size, void* d_ws, size_t ws_size,
                              hipStream_t stream) {
    const float* img   = (const float*)d_in[0];   // [16384, 512] f32
    const float* txt   = (const float*)d_in[1];   // [16384, 512] f32
    const float* scale = (const float*)d_in[2];   // scalar f32
    // d_in[3] = labels (int64) -- unused by the reference computation
    float* out = (float*)d_out;

    char* ws = (char*)d_ws;
    _Float16* imgh = (_Float16*)ws;                                        // 16 MB
    _Float16* txth = (_Float16*)(ws + (size_t)N_FEAT * D_FEAT * 2);        // 16 MB
    float* rowsum  = (float*)  (ws + (size_t)N_FEAT * D_FEAT * 4);         // 64 KB
    float* colsum  = rowsum + N_FEAT;                                      // 64 KB
    float* diag    = colsum + N_FEAT;                                      // 64 KB

    const int nElem = N_FEAT * D_FEAT;
    cvt_f32_to_f16_kernel<<<2048, 256, 0, stream>>>(img, imgh, nElem);
    cvt_f32_to_f16_kernel<<<2048, 256, 0, stream>>>(txt, txth, nElem);
    zero_f32_kernel<<<(2 * N_FEAT + 255) / 256, 256, 0, stream>>>(rowsum, 2 * N_FEAT);

    dim3 grid(N_FEAT / 128, N_FEAT / 128);  // 128 x 128 blocks
    infonce_tile_kernel<<<grid, 256, 0, stream>>>(imgh, txth, scale,
                                                  rowsum, colsum, diag);
    infonce_finalize_kernel<<<1, 256, 0, stream>>>(rowsum, colsum, diag, out);
}
